// ScaledDotProductAttention_40707700032213
// MI455X (gfx1250) — matrix operand order repaired and checked by reading, not compiled, not run on hardware
//
#include <hip/hip_runtime.h>

typedef __attribute__((ext_vector_type(16))) _Float16 v16h;
typedef __attribute__((ext_vector_type(8)))  _Float16 v8h;
typedef __attribute__((ext_vector_type(8)))  float    v8f;
typedef __attribute__((ext_vector_type(4)))  int      v4i;

#define L_SEQ   4096
#define DMODEL  1024
#define NHEAD   16
#define HD      64
#define BM      128            // query rows per block (8 waves x 16)
#define BN      32             // key rows per iteration
#define PB_PAD  (BN + 16)      // 48 halves -> 96B row stride
#define QSCALE  (0.125f * 1.44269504088896340736f)   // 1/sqrt(64) * log2(e)

#ifndef __has_builtin
#define __has_builtin(x) 0
#endif

// ---- async global->LDS copy (ASYNCcnt path) ------------------------------
#if __has_builtin(__builtin_amdgcn_global_load_async_to_lds_b128)
typedef __attribute__((address_space(1))) v4i g1v4i;
typedef __attribute__((address_space(3))) v4i l3v4i;
#define ASYNC_COPY_B128(ldsp, gp)                                   \
    __builtin_amdgcn_global_load_async_to_lds_b128(                 \
        (g1v4i*)(unsigned long long)(gp),                           \
        (l3v4i*)(unsigned)(unsigned long long)(ldsp), 0, 0)
#else
#define ASYNC_COPY_B128(ldsp, gp) \
    asm volatile("global_load_async_to_lds_b128 %0, %1, off" \
                 :: "v"((unsigned)(unsigned long long)(ldsp)), \
                    "v"((unsigned long long)(gp)) : "memory")
#endif

#if __has_builtin(__builtin_amdgcn_s_wait_asynccnt)
#define WAIT_ASYNC(n) __builtin_amdgcn_s_wait_asynccnt(n)
#else
#define WAIT_ASYNC(n) asm volatile("s_wait_asynccnt " #n ::: "memory")
#endif

// ---- raw VALU max (no canonicalize) and fused DPP butterfly steps --------
__device__ __forceinline__ float fmax_raw(float a, float b) {
    float d;
    asm("v_max_num_f32_e32 %0, %1, %2" : "=v"(d) : "v"(a), "v"(b));
    return d;
}
#define ROWMAX_STEP(x, XM)                                                        \
    asm("v_max_num_f32_dpp %0, %1, %0 " XM                                        \
        " row_mask:0xf bank_mask:0xf bound_ctrl:1"                                \
        : "+v"(x) : "v"(x))
__device__ __forceinline__ float rowmax16(float x) {
    ROWMAX_STEP(x, "row_xmask:8");
    ROWMAX_STEP(x, "row_xmask:4");
    ROWMAX_STEP(x, "row_xmask:2");
    ROWMAX_STEP(x, "row_xmask:1");
    return x;
}

// ===========================================================================
// Prepass 1: Kth[h][d][n] = (f16) K[n][h*64+d]   (per-head transposed K)
// ===========================================================================
__global__ __launch_bounds__(256)
void conv_k_transpose(const float* __restrict__ K, _Float16* __restrict__ Kth) {
    __shared__ _Float16 tile[64][64 + 8];
    const int h  = blockIdx.x % NHEAD;
    const int n0 = (blockIdx.x / NHEAD) * 64;
    const int t  = threadIdx.x;
    {
        const int r   = t >> 2;
        const int c16 = (t & 3) * 16;
        const float* src = K + (size_t)(n0 + r) * DMODEL + h * HD + c16;
#pragma unroll
        for (int i = 0; i < 16; ++i) tile[r][c16 + i] = (_Float16)src[i];
    }
    __syncthreads();
    {
        const int d  = t >> 2;
        const int nn = (t & 3) * 16;
        v8h a, b;
#pragma unroll
        for (int i = 0; i < 8; ++i) { a[i] = tile[nn + i][d]; b[i] = tile[nn + 8 + i][d]; }
        _Float16* dst = Kth + ((size_t)h * HD + d) * L_SEQ + n0 + nn;
        *(v8h*)dst = a;
        *(v8h*)(dst + 8) = b;
    }
}

// ===========================================================================
// Prepass 2: Vh[h][n][d] = (f16) V[n][h*64+d]    (head-major V)
// ===========================================================================
__global__ __launch_bounds__(256)
void conv_v_f16(const float* __restrict__ V, _Float16* __restrict__ Vh) {
    const size_t t  = (size_t)blockIdx.x * 256 + threadIdx.x;
    const int n  = (int)(t / (DMODEL / 8));
    const int c8 = (int)(t % (DMODEL / 8)) * 8;
    const int h  = c8 / HD, d = c8 % HD;
    const float* src = V + (size_t)n * DMODEL + c8;
    v8h o;
#pragma unroll
    for (int i = 0; i < 8; ++i) o[i] = (_Float16)src[i];
    *(v8h*)&Vh[((size_t)h * L_SEQ + n) * HD + d] = o;
}

// ===========================================================================
// Flash attention: double-buffered async K/V tiles, f16 WMMA, DPP softmax
// ===========================================================================
__global__ __launch_bounds__(256, 1)
void fa_fwd_f16wmma(const float* __restrict__ Q,
                    const _Float16* __restrict__ Kth,
                    const _Float16* __restrict__ Vh,
                    float* __restrict__ Out)
{
    __shared__ __align__(32) _Float16 KtBuf[2][HD][BN];
    __shared__ __align__(32) _Float16 VsBuf[2][BN][HD];
    __shared__ __align__(32) _Float16 Pb[8][16][PB_PAD];

    const int tid   = threadIdx.x;
    const int wave  = tid >> 5;
    const int lane  = tid & 31;
    const int l15   = lane & 15;
    const int lhalf = lane >> 4;

    const int mtile = blockIdx.x / NHEAD;
    const int head  = blockIdx.x % NHEAD;
    const int row0  = mtile * BM;

    const _Float16* KthH = Kth + (size_t)head * HD * L_SEQ;
    const _Float16* VhH  = Vh  + (size_t)head * L_SEQ * HD;

    const int qrow = row0 + wave * 16 + l15;
    const float* qp = Q + (size_t)qrow * DMODEL + head * HD;
    v16h qa[2];
#pragma unroll
    for (int f = 0; f < 2; ++f) {
        const int kb = f * 32 + lhalf * 8;
#pragma unroll
        for (int i = 0; i < 8; ++i) {
            qa[f][i]     = (_Float16)(qp[kb + i]      * QSCALE);
            qa[f][8 + i] = (_Float16)(qp[kb + 16 + i] * QSCALE);
        }
    }
    v16h ones;
#pragma unroll
    for (int i = 0; i < 16; ++i) ones[i] = (_Float16)1.0f;

    v8f o0 = {}, o1 = {}, o2 = {}, o3 = {};
    float mi[8], li[8];
#pragma unroll
    for (int v = 0; v < 8; ++v) { mi[v] = -__builtin_inff(); li[v] = 0.0f; }

    const int kd  = tid >> 2;
    const int kob = (tid & 3) * 8;
    const int vn  = tid >> 3;
    const int vob = (tid & 7) * 8;

#define ISSUE_TILE(JT, BUF)                                                           \
    do {                                                                              \
        const int kb_ = (JT) * BN;                                                    \
        ASYNC_COPY_B128(&KtBuf[BUF][kd][kob], KthH + (size_t)kd * L_SEQ + kb_ + kob); \
        ASYNC_COPY_B128(&VsBuf[BUF][vn][vob], VhH + (size_t)(kb_ + vn) * HD + vob);   \
    } while (0)

    const int NT = L_SEQ / BN;
    ISSUE_TILE(0, 0);

    for (int jt = 0; jt < NT; ++jt) {
        const int cb = jt & 1;
        if (jt + 1 < NT) { ISSUE_TILE(jt + 1, cb ^ 1); WAIT_ASYNC(2); }
        else             { WAIT_ASYNC(0); }
        __syncthreads();

        // ---- hoisted B-fragments ----
        v16h bk00, bk01, bk10, bk11, bv0, bv1, bv2, bv3;
#pragma unroll
        for (int s = 0; s < 16; ++s) {
            const int kk = 8 * lhalf + (s < 8 ? s : s + 8);
            bk00[s] = KtBuf[cb][kk][l15];
            bk01[s] = KtBuf[cb][kk][16 + l15];
            bk10[s] = KtBuf[cb][32 + kk][l15];
            bk11[s] = KtBuf[cb][32 + kk][16 + l15];
            bv0[s]  = VsBuf[cb][kk][l15];
            bv1[s]  = VsBuf[cb][kk][16 + l15];
            bv2[s]  = VsBuf[cb][kk][32 + l15];
            bv3[s]  = VsBuf[cb][kk][48 + l15];
        }

        // ---- S = Q K^T, f-outer, reuse_a on repeated-A WMMAs ----
        v8f c0 = {}, c1 = {};
        c0 = __builtin_amdgcn_wmma_f32_16x16x32_f16(false, qa[0], false, bk00, (short)0, c0, false, false);
        c1 = __builtin_amdgcn_wmma_f32_16x16x32_f16(false, qa[0], false, bk01, (short)0, c1, true,  false);
        c0 = __builtin_amdgcn_wmma_f32_16x16x32_f16(false, qa[1], false, bk10, (short)0, c0, false, false);
        c1 = __builtin_amdgcn_wmma_f32_16x16x32_f16(false, qa[1], false, bk11, (short)0, c1, true,  false);

        // ---- online softmax (exp2 domain), DPP row max, stage P ----
        float alpha8[8];
#pragma unroll
        for (int v = 0; v < 8; ++v) {
            const float s0 = c0[v];
            const float s1 = c1[v];
            const float rm = rowmax16(fmax_raw(s0, s1));
            const float mn = fmax_raw(mi[v], rm);
            const float a  = exp2f(mi[v] - mn);
            const float p0 = exp2f(s0 - mn);
            const float p1 = exp2f(s1 - mn);
            mi[v] = mn;
            alpha8[v] = a;
            const int prow = v + 8 * lhalf;
            Pb[wave][prow][l15]      = (_Float16)p0;
            Pb[wave][prow][16 + l15] = (_Float16)p1;
        }
        // rescale accumulators by alpha (per row)
#pragma unroll
        for (int v = 0; v < 8; ++v) {
            o0[v] *= alpha8[v]; o1[v] *= alpha8[v];
            o2[v] *= alpha8[v]; o3[v] *= alpha8[v];
        }

        // ---- P as A-fragment from LDS ----
        v16h pa;
        {
            const v8h lo = *(const v8h*)&Pb[wave][l15][lhalf * 8];
            const v8h hi = *(const v8h*)&Pb[wave][l15][lhalf * 8 + 16];
#pragma unroll
            for (int i = 0; i < 8; ++i) { pa[i] = lo[i]; pa[8 + i] = hi[i]; }
        }

        // ---- rsum + O += P*V, all sharing A=pa (reuse_a chain) ----
        v8f rsum = {};
        rsum = __builtin_amdgcn_wmma_f32_16x16x32_f16(false, pa, false, ones, (short)0, rsum, false, false);
        o0   = __builtin_amdgcn_wmma_f32_16x16x32_f16(false, pa, false, bv0,  (short)0, o0,   true,  false);
        o1   = __builtin_amdgcn_wmma_f32_16x16x32_f16(false, pa, false, bv1,  (short)0, o1,   true,  false);
        o2   = __builtin_amdgcn_wmma_f32_16x16x32_f16(false, pa, false, bv2,  (short)0, o2,   true,  false);
        o3   = __builtin_amdgcn_wmma_f32_16x16x32_f16(false, pa, false, bv3,  (short)0, o3,   true,  false);

#pragma unroll
        for (int v = 0; v < 8; ++v) li[v] = li[v] * alpha8[v] + rsum[v];
        __syncthreads();
    }

    // ---- epilogue: O / l ----
    const int orow_base = row0 + wave * 16;
#pragma unroll
    for (int v = 0; v < 8; ++v) {
        const float inv = 1.0f / li[v];
        const size_t r = (size_t)(orow_base + v + 8 * lhalf) * DMODEL + head * HD + l15;
        Out[r]      = o0[v] * inv;
        Out[r + 16] = o1[v] * inv;
        Out[r + 32] = o2[v] * inv;
        Out[r + 48] = o3[v] * inv;
    }
#undef ISSUE_TILE
}

extern "C" void kernel_launch(void* const* d_in, const int* in_sizes, int n_in,
                              void* d_out, int out_size, void* d_ws, size_t ws_size,
                              hipStream_t stream) {
    (void)in_sizes; (void)n_in; (void)out_size; (void)ws_size;  // needs >= 16 MB ws
    const float* Q = (const float*)d_in[0];
    const float* K = (const float*)d_in[1];
    const float* V = (const float*)d_in[2];
    float* Out = (float*)d_out;

    _Float16* Kth = (_Float16*)d_ws;                                   // 8 MB
    _Float16* Vh  = (_Float16*)((char*)d_ws + (size_t)NHEAD * HD * L_SEQ * 2);

    conv_k_transpose<<<NHEAD * (L_SEQ / 64), 256, 0, stream>>>(K, Kth);
    conv_v_f16<<<(L_SEQ * DMODEL / 8) / 256, 256, 0, stream>>>(V, Vh);
    fa_fwd_f16wmma<<<(L_SEQ / BM) * NHEAD, 256, 0, stream>>>(Q, Kth, Vh, Out);
}
